// SimpleSSMLayer_65627100283162
// MI455X (gfx1250) — compile-verified, matrix operand order checked by reading, not run on hardware
//
#include <hip/hip_runtime.h>
#include <math.h>

#define B_SZ   32
#define SEQ_L  4096
#define DMODEL 512
#define SLICES 4
#define SW     128   // scan slice width (columns per workgroup)

typedef __attribute__((ext_vector_type(16))) _Float16 v16h;
typedef __attribute__((ext_vector_type(8)))  float    v8f;

// ---------------------------------------------------------------------------
// gfx1250 async global->LDS copy (per-lane 16B), tracked by ASYNCcnt.
// ---------------------------------------------------------------------------
__device__ __forceinline__ void async_ld_b128(const void* sbase, unsigned goff_bytes,
                                              void* lds_dst)
{
    unsigned lds_off = (unsigned)(unsigned long long)(uintptr_t)lds_dst;
    asm volatile("global_load_async_to_lds_b128 %0, %1, %2"
                 :: "v"(lds_off), "v"(goff_bytes), "s"(sbase) : "memory");
}
// wait until at most N async ops outstanding (in-order completion per wave)
__device__ __forceinline__ void wait_async_le2()
{
    asm volatile("s_wait_asynccnt 0x2" ::: "memory");
}
__device__ __forceinline__ void wait_async_le0()
{
    asm volatile("s_wait_asynccnt 0x0" ::: "memory");
}

__device__ __forceinline__ float fast_silu(float v)
{
    return v * __builtin_amdgcn_rcpf(1.0f + __expf(-v));
}
__device__ __forceinline__ float fast_tanh(float x)
{
    // 1 - 2/(e^{2x}+1); saturates correctly at +-1 for large |x|
    float e = __expf(2.0f * x);
    return 1.0f - 2.0f * __builtin_amdgcn_rcpf(e + 1.0f);
}

// ---------------------------------------------------------------------------
// Prep: f32 -> f16 bulk convert (x) and weight transpose-to-f16 ([k][n]->[n][k])
// ---------------------------------------------------------------------------
__global__ __launch_bounds__(256)
void cvt_f16_kernel(const float* __restrict__ in, _Float16* __restrict__ out, size_t n)
{
    size_t i = ((size_t)blockIdx.x * 256 + threadIdx.x) * 4;
    if (i + 3 < n) {
        float4 v = *(const float4*)(in + i);
        out[i + 0] = (_Float16)v.x;
        out[i + 1] = (_Float16)v.y;
        out[i + 2] = (_Float16)v.z;
        out[i + 3] = (_Float16)v.w;
    }
}

__global__ __launch_bounds__(256)
void transpose_f16_kernel(const float* __restrict__ in, _Float16* __restrict__ out,
                          int K, int N)   // in: [K][N] f32, out: [N][K] f16
{
    int t = blockIdx.x * 256 + threadIdx.x;
    if (t < N * K) {
        int nn = t / K, kk = t - nn * K;
        out[t] = (_Float16)in[(size_t)kk * N + nn];
    }
}

// ---------------------------------------------------------------------------
// WMMA-tiled GEMM: out = A[M,K] @ B^T[N,K] (+bias). f16 in, f32 WMMA accum.
// Double-buffered tiles streamed into LDS with global_load_async_to_lds_b128;
// prefetch of tile k+1 overlaps WMMA on tile k (wait ASYNCcnt<=2 keeps only
// the in-flight prefetch outstanding).
// Block = 256 threads (8 waves), tile 64x64, BK=32.
// MODE 0: N==1024 projection; col<512 -> f16 out0 (x_ssm), col>=512 -> silu f16 out1.
// MODE 1: f16 plain store to out0 (no bias).
// MODE 2: f32 store to out0 (+bias).
// ---------------------------------------------------------------------------
template <int MODE>
__global__ __launch_bounds__(256)
void wmma_gemm_kernel(const _Float16* __restrict__ Ag, const _Float16* __restrict__ BgT,
                      const float* __restrict__ bias,
                      void* __restrict__ out0, void* __restrict__ out1,
                      int M, int N, int K)
{
    __shared__ __align__(16) _Float16 Ah[2][64][32];   // A tile [m][k]
    __shared__ __align__(16) _Float16 Bt[2][64][32];   // B tile [n][k]

    const int tid  = threadIdx.x;
    const int lane = tid & 31;
    const int wave = tid >> 5;
    const int m0   = blockIdx.y * 64;
    const int n0   = blockIdx.x * 64;

    const int mt = wave & 3;          // 16-row band within 64
    const int nb = (wave >> 2) * 32;  // 32-col band: 2 adjacent 16x16 tiles

    const int r  = lane & 15;
    const int hi = lane >> 4;
    const int kb = hi * 8;            // K phase per ISA 16-bit A/B layout

    // each thread owns one 16B chunk of each tile: 64 rows x 4 chunks
    const int trow = tid >> 2;        // 0..63
    const int tq   = (tid & 3) * 8;   // half-offset within row

    const size_t rowA = (size_t)(m0 + trow) * K + tq;  // element offsets
    const size_t rowB = (size_t)(n0 + trow) * K + tq;

    v8f c0 = {}; v8f c1 = {};

    // prologue: prefetch tile 0 into buffer 0
    async_ld_b128(Ag,  (unsigned)(rowA * 2), &Ah[0][trow][tq]);
    async_ld_b128(BgT, (unsigned)(rowB * 2), &Bt[0][trow][tq]);

    for (int k0 = 0; k0 < K; k0 += 32) {
        const int buf = (k0 >> 5) & 1;
        if (k0 + 32 < K) {
            async_ld_b128(Ag,  (unsigned)((rowA + k0 + 32) * 2), &Ah[buf ^ 1][trow][tq]);
            async_ld_b128(BgT, (unsigned)((rowB + k0 + 32) * 2), &Bt[buf ^ 1][trow][tq]);
            wait_async_le2();   // current tile's copies retired; prefetch in flight
        } else {
            wait_async_le0();
        }
        __syncthreads();

        v16h af, bf0, bf1;
        #pragma unroll
        for (int h = 0; h < 16; ++h) {
            int kk = kb + (h < 8 ? h : h + 8);
            af[h]  = Ah[buf][mt * 16 + r][kk];
            bf0[h] = Bt[buf][nb + r][kk];
            bf1[h] = Bt[buf][nb + 16 + r][kk];
        }
        c0 = __builtin_amdgcn_wmma_f32_16x16x32_f16(false, af, false, bf0,
                                                    (short)0, c0, false, false);
        c1 = __builtin_amdgcn_wmma_f32_16x16x32_f16(false, af, false, bf1,
                                                    (short)0, c1, false, false);
        __syncthreads();   // readers done before this buffer is overwritten
    }

    // epilogue: D element (m = i + 8*hi, n = r) per ISA C/D layout
    const int col0 = n0 + nb + r;
    const int col1 = col0 + 16;
    float b0 = 0.0f, b1 = 0.0f;
    if (MODE != 1) { b0 = bias[col0]; b1 = bias[col1]; }

    #pragma unroll
    for (int i = 0; i < 8; ++i) {
        int row = m0 + mt * 16 + i + 8 * hi;
        float v0 = c0[i] + b0;
        float v1 = c1[i] + b1;
        if (MODE == 0) {
            if (col0 < DMODEL) {   // block-uniform: 64-wide bands never straddle 512
                ((_Float16*)out0)[(size_t)row * DMODEL + col0] = (_Float16)v0;
                ((_Float16*)out0)[(size_t)row * DMODEL + col1] = (_Float16)v1;
            } else {
                ((_Float16*)out1)[(size_t)row * DMODEL + (col0 - DMODEL)] = (_Float16)fast_silu(v0);
                ((_Float16*)out1)[(size_t)row * DMODEL + (col1 - DMODEL)] = (_Float16)fast_silu(v1);
            }
        } else if (MODE == 1) {
            ((_Float16*)out0)[(size_t)row * N + col0] = (_Float16)v0;
            ((_Float16*)out0)[(size_t)row * N + col1] = (_Float16)v1;
        } else {
            ((float*)out0)[(size_t)row * N + col0] = v0;
            ((float*)out0)[(size_t)row * N + col1] = v1;
        }
    }
}

// ---------------------------------------------------------------------------
// Sequential SSM scan. grid = (SLICES, B_SZ), block = 256 threads.
// A-row slice + Cm-column slice pinned in LDS as f16 (256KB of the 320KB WGP
// LDS). 2KB h-vector exchanged through L2 with agent-scope atomics; slices
// sync via a monotonic counter barrier. gated aliases xB (safe: consumed first).
// ---------------------------------------------------------------------------
__global__ __launch_bounds__(256)
void ssm_scan_kernel(const _Float16* xB, const _Float16* __restrict__ xssm,
                     const _Float16* __restrict__ gate,
                     const float* __restrict__ Amat, const float* __restrict__ Cmat,
                     const float* __restrict__ Dvec,
                     _Float16* gated, float* hbuf, unsigned* ctr, int L)
{
    __shared__ _Float16 Ah[SW][DMODEL];   // A[n0+n][d]
    __shared__ _Float16 Ct[SW][DMODEL];   // Cm[d][n0+n] transposed
    __shared__ float    hsh[DMODEL];

    const int b   = blockIdx.y;
    const int n0  = blockIdx.x * SW;
    const int tid = threadIdx.x;

    for (int i = tid; i < SW * DMODEL; i += 256) {
        int n = i >> 9, d = i & (DMODEL - 1);
        Ah[n][d] = (_Float16)Amat[(size_t)(n0 + n) * DMODEL + d];
        Ct[n][d] = (_Float16)Cmat[(size_t)d * DMODEL + (n0 + n)];
    }
    hsh[tid] = 0.0f; hsh[tid + 256] = 0.0f;
    __syncthreads();

    const int n    = tid >> 1;          // 0..127: output column within slice
    const int half = tid & 1;           // 2 threads split the K=512 reduction
    const int koff = half * 256;
    unsigned* myctr = ctr + b;
    const float dv = Dvec[n0 + n];

    for (int t = 0; t < L; ++t) {
        const size_t base = ((size_t)b * L + t) * DMODEL + n0 + n;

        // z[n] = sum_d h[d] * A[n0+n][d]   (h @ A^T)
        float acc = 0.0f;
        #pragma unroll 8
        for (int k = 0; k < 256; ++k)
            acc += (float)Ah[n][koff + k] * hsh[koff + k];
        acc += __shfl_xor(acc, 1);
        if (half == 0) {
            float hn = fast_tanh(acc + (float)xB[base]);
            __hip_atomic_store(&hbuf[b * DMODEL + n0 + n], hn,
                               __ATOMIC_RELEASE, __HIP_MEMORY_SCOPE_AGENT);
        }

        // barrier phase 1: all slices published h_new(t)
        __syncthreads();
        if (tid == 0) {
            __hip_atomic_fetch_add(myctr, 1u, __ATOMIC_ACQ_REL, __HIP_MEMORY_SCOPE_AGENT);
            unsigned tgt = (unsigned)(2 * t + 1) * SLICES;
            while (__hip_atomic_load(myctr, __ATOMIC_ACQUIRE, __HIP_MEMORY_SCOPE_AGENT) < tgt)
                __builtin_amdgcn_s_sleep(1);
        }
        __syncthreads();

        // pull full h_new into LDS (agent-scope: bypass stale WGP$)
        hsh[tid] = __hip_atomic_load(&hbuf[b * DMODEL + tid],
                                     __ATOMIC_ACQUIRE, __HIP_MEMORY_SCOPE_AGENT);
        hsh[tid + 256] = __hip_atomic_load(&hbuf[b * DMODEL + tid + 256],
                                           __ATOMIC_ACQUIRE, __HIP_MEMORY_SCOPE_AGENT);
        __syncthreads();

        // barrier phase 2: all slices consumed h(t) before anyone writes h(t+1)
        if (tid == 0) {
            __hip_atomic_fetch_add(myctr, 1u, __ATOMIC_ACQ_REL, __HIP_MEMORY_SCOPE_AGENT);
            unsigned tgt = (unsigned)(2 * t + 2) * SLICES;
            while (__hip_atomic_load(myctr, __ATOMIC_ACQUIRE, __HIP_MEMORY_SCOPE_AGENT) < tgt)
                __builtin_amdgcn_s_sleep(1);
        }
        __syncthreads();

        // y[n] = h_new . Cm[:,n0+n] + Dv*x ; fuse gating
        float yacc = 0.0f;
        #pragma unroll 8
        for (int k = 0; k < 256; ++k)
            yacc += (float)Ct[n][koff + k] * hsh[koff + k];
        yacc += __shfl_xor(yacc, 1);
        if (half == 0) {
            float y = yacc + dv * (float)xssm[base];
            gated[base] = (_Float16)(y * (float)gate[base]);
        }
    }
}

// ---------------------------------------------------------------------------
extern "C" void kernel_launch(void* const* d_in, const int* in_sizes, int n_in,
                              void* d_out, int out_size, void* d_ws, size_t ws_size,
                              hipStream_t stream)
{
    (void)in_sizes; (void)n_in; (void)out_size; (void)ws_size;
    const float* x     = (const float*)d_in[0];
    const float* W_in  = (const float*)d_in[1];
    const float* b_in  = (const float*)d_in[2];
    const float* W_out = (const float*)d_in[3];
    const float* b_out = (const float*)d_in[4];
    const float* A     = (const float*)d_in[5];
    const float* Bm    = (const float*)d_in[6];
    const float* Cm    = (const float*)d_in[7];
    const float* Dv    = (const float*)d_in[8];

    const int M = B_SZ * SEQ_L;                       // 131072 rows
    const size_t nBLD = (size_t)M * DMODEL;           // 67,108,864 elements
    char* ws = (char*)d_ws;
    _Float16* xh      = (_Float16*)ws;                          // 128MB
    _Float16* xssm_h  = (_Float16*)(ws + 1 * nBLD * 2);         // 128MB
    _Float16* gate_h  = (_Float16*)(ws + 2 * nBLD * 2);         // 128MB
    _Float16* xB_h    = (_Float16*)(ws + 3 * nBLD * 2);         // 128MB, reused as gated
    char*     wsw     = ws + 4 * nBLD * 2;
    _Float16* WinT_h  = (_Float16*)wsw;                         // 1024x512 f16 = 1MB
    _Float16* BmT_h   = (_Float16*)(wsw + (size_t)1024 * 512 * 2);
    _Float16* WoutT_h = (_Float16*)(wsw + (size_t)1024 * 512 * 2 + (size_t)512 * 512 * 2);
    char*     wse     = wsw + (size_t)1024 * 512 * 2 + 2 * (size_t)512 * 512 * 2;
    float*    hbuf    = (float*)wse;                            // 64KB
    unsigned* ctr     = (unsigned*)(wse + (size_t)B_SZ * DMODEL * 4);

    hipMemsetAsync(ctr, 0, B_SZ * sizeof(unsigned), stream);

    // prep: x -> f16 ; weights -> transposed f16
    cvt_f16_kernel<<<(unsigned)(nBLD / 1024), 256, 0, stream>>>(x, xh, nBLD);
    transpose_f16_kernel<<<(1024 * 512) / 256, 256, 0, stream>>>(W_in, WinT_h, DMODEL, 1024);
    transpose_f16_kernel<<<(512 * 512) / 256, 256, 0, stream>>>(Bm, BmT_h, DMODEL, DMODEL);
    transpose_f16_kernel<<<(512 * 512) / 256, 256, 0, stream>>>(W_out, WoutT_h, DMODEL, DMODEL);

    // 1) x_proj = x @ W_in + b_in -> x_ssm | silu(x_gate)
    wmma_gemm_kernel<0><<<dim3(1024 / 64, M / 64), 256, 0, stream>>>(
        xh, WinT_h, b_in, xssm_h, gate_h, M, 1024, DMODEL);

    // 2) xB = x_ssm @ Bm
    wmma_gemm_kernel<1><<<dim3(DMODEL / 64, M / 64), 256, 0, stream>>>(
        xssm_h, BmT_h, nullptr, xB_h, nullptr, M, DMODEL, DMODEL);

    // 3) sequential scan + gating (gated written in-place over xB)
    ssm_scan_kernel<<<dim3(SLICES, B_SZ), 256, 0, stream>>>(
        xB_h, xssm_h, gate_h, A, Cm, Dv, xB_h, hbuf, ctr, SEQ_L);

    // 4) out = gated @ W_out + b_out  (f32 output)
    wmma_gemm_kernel<2><<<dim3(DMODEL / 64, M / 64), 256, 0, stream>>>(
        xB_h, WoutT_h, b_out, d_out, nullptr, M, DMODEL, DMODEL);
}
